// ConformerMHSARelPosV1_81071802679903
// MI455X (gfx1250) — matrix operand order repaired and checked by reading, not compiled, not run on hardware
//
#include <hip/hip_runtime.h>
#include <hip/hip_bf16.h>

// ---- problem constants -------------------------------------------------
#define BB 8
#define TT 1024
#define EE 512
#define HH 8
#define DHH 64
#define MM (BB * TT)        // 8192 rows for all the big GEMMs

typedef __bf16 bf16_t;
typedef __attribute__((ext_vector_type(16))) __bf16 bf16x16;
typedef __attribute__((ext_vector_type(8)))  float  f32x8;
typedef __attribute__((ext_vector_type(4)))  unsigned int u32x4;
typedef __attribute__((ext_vector_type(8)))  int i32x8;
typedef __attribute__((ext_vector_type(4)))  int i32x4;

union FragU { bf16x16 v; uint4 q[2]; };

// A-matrix 16x32 bf16 fragment (row-major source, K contiguous).
// ISA layout: lanes 0-15 -> M=lane, K {0..7,16..23}; lanes 16-31 -> M=lane-16, K {8..15,24..31}
__device__ __forceinline__ bf16x16 load_frag_a(const bf16_t* base, int stride, int lane) {
    int r    = lane & 15;
    int koff = (lane >> 4) * 8;
    FragU f;
    const bf16_t* p = base + (size_t)r * stride + koff;
    f.q[0] = *reinterpret_cast<const uint4*>(p);
    f.q[1] = *reinterpret_cast<const uint4*>(p + 16);
    return f.v;
}

// B-matrix 32x16 bf16 fragment. Source stored "N-major" (each N row has K contiguous).
// Layout: lane -> column N=lane&15, K {0..7,16..23} (lanes 0-15) / {8..15,24..31} (lanes 16-31)
__device__ __forceinline__ bf16x16 load_frag_b(const bf16_t* base, int stride, int lane) {
    int n    = lane & 15;
    int koff = (lane >> 4) * 8;
    FragU f;
    const bf16_t* p = base + (size_t)n * stride + koff;
    f.q[0] = *reinterpret_cast<const uint4*>(p);
    f.q[1] = *reinterpret_cast<const uint4*>(p + 16);
    return f.v;
}

// ---- Tensor Data Mover: DMA an 8-row x 128-col bf16 slice of a 512x512
// row-major matrix (stride 512 elements) into LDS. Per CDNA5 ISA §8 D# layout.
__device__ __forceinline__ void tdm_load_slice(const bf16_t* gsrc, bf16_t* ldst) {
    unsigned long long ga = (unsigned long long)(const void*)gsrc;
    unsigned int alo = (unsigned int)ga;
    unsigned int ahi = (unsigned int)(ga >> 32);
    unsigned int lds = (unsigned int)(unsigned long long)(void*)ldst; // low 32b = LDS byte offset
    alo = (unsigned int)__builtin_amdgcn_readfirstlane((int)alo);
    ahi = (unsigned int)__builtin_amdgcn_readfirstlane((int)ahi);
    lds = (unsigned int)__builtin_amdgcn_readfirstlane((int)lds);

    // group0: count=1 | lds_addr | global_addr[56:0] | type=2
    u32x4 g0 = { 1u, lds, alo, (ahi & 0x01FFFFFFu) | 0x80000000u };
    // group1: data_size=2B; tensor_dim0=512, tensor_dim1=512;
    //         tile_dim0=128 (K), tile_dim1=8 (rows); tensor_dim0_stride=512
    i32x8 g1 = { (int)0x00010000u,        // [17:16] data_size=1 (2 bytes)
                 (int)(512u << 16),       // tensor_dim0[15:0] @ bits[63:48]
                 (int)(512u << 16),       // tensor_dim0 hi=0 | tensor_dim1[15:0]
                 (int)(128u << 16),       // tensor_dim1 hi=0 | tile_dim0=128
                 8,                       // tile_dim1=8, tile_dim2=0
                 512,                     // tensor_dim0_stride (48b, low 32)
                 0, 0 };                  // stride hi, tensor_dim1_stride=0
    i32x4 z4 = { 0, 0, 0, 0 };
#if __clang_major__ >= 23
    i32x8 z8 = { 0, 0, 0, 0, 0, 0, 0, 0 };
    __builtin_amdgcn_tensor_load_to_lds(g0, g1, z4, z4, z8, 0);
#else
    __builtin_amdgcn_tensor_load_to_lds(g0, g1, z4, z4, 0);
#endif
}

// ---- kernel 1: f32 -> bf16 weight convert ------------------------------
__global__ void f2bf_kernel(const float* __restrict__ src, bf16_t* __restrict__ dst, int n) {
    int i = blockIdx.x * blockDim.x + threadIdx.x;
    if (i < n) dst[i] = (bf16_t)src[i];
}

// ---- kernel 2: LayerNorm -> bf16, one wave32 per row -------------------
__global__ __launch_bounds__(256) void ln_kernel(const float* __restrict__ x,
                                                 const float* __restrict__ g,
                                                 const float* __restrict__ bta,
                                                 bf16_t* __restrict__ xn) {
    int lane = threadIdx.x & 31;
    int row  = blockIdx.x * 8 + (threadIdx.x >> 5);
    const float* xr = x + (size_t)row * EE;
    float v[16];
    float s = 0.f;
#pragma unroll
    for (int i = 0; i < 16; ++i) { v[i] = xr[lane + 32 * i]; s += v[i]; }
#pragma unroll
    for (int m = 1; m < 32; m <<= 1) s += __shfl_xor(s, m, 32);
    float mu = s * (1.0f / EE);
    float q = 0.f;
#pragma unroll
    for (int i = 0; i < 16; ++i) { float d = v[i] - mu; q += d * d; }
#pragma unroll
    for (int m = 1; m < 32; m <<= 1) q += __shfl_xor(q, m, 32);
    float rinv = rsqrtf(q * (1.0f / EE) + 1e-5f);
    bf16_t* out = xn + (size_t)row * EE;
#pragma unroll
    for (int i = 0; i < 16; ++i) {
        int c = lane + 32 * i;
        out[c] = (bf16_t)((v[i] - mu) * rinv * g[c] + bta[c]);
    }
}

// ---- kernel 3: trig tables trig[t*64+f] = f<32 ? cos(t*w_f) : sin(t*w_f)
__global__ void trig_kernel(float* __restrict__ trig) {
    int idx = blockIdx.x * blockDim.x + threadIdx.x;   // TT*64
    int t = idx >> 6, f = idx & 63, gidx = f & 31;
    float w = __expf(-__logf(10000.f) * ((float)gidx / 32.f));
    float a = (float)t * w;
    trig[idx] = (f < 32) ? __cosf(a) : __sinf(a);
}

// ---- kernel 4: bf16 WMMA GEMM with TDM-staged weight tiles -------------
// out = A(M x 512) @ W(512 x 512)^T + bias
// mode 0: -> Qtmp  bf16 [m,n]
// mode 1: -> Khat  bf16 [b,h,t, n%64]
// mode 2: -> Vt    bf16 [b,h, n%64, t]
// mode 3: -> d_out f32  [m,n]
__global__ __launch_bounds__(256)
void gemm_kernel(const bf16_t* __restrict__ A, const bf16_t* __restrict__ W,
                 const float* __restrict__ bias, int mode, void* __restrict__ out) {
    __shared__ __align__(16) bf16_t btile[2][64 * 128];   // double-buffered 16KB tiles
    int lane = threadIdx.x & 31;
    int wave = threadIdx.x >> 5;
    int m0 = blockIdx.x * 128 + wave * 16;
    int n0 = blockIdx.y * 64;

    // Each wave DMAs its own 8 N-rows of the 64x128 weight chunk via the TDM.
    const bf16_t* wsrc = W + (size_t)(n0 + wave * 8) * EE;
    tdm_load_slice(wsrc, &btile[0][wave * 8 * 128]);

    f32x8 z = {0.f,0.f,0.f,0.f,0.f,0.f,0.f,0.f};
    f32x8 acc[4] = {z, z, z, z};
    const bf16_t* arow = A + (size_t)m0 * EE;

#pragma unroll
    for (int kc = 0; kc < 4; ++kc) {          // K chunks of 128
        int kb0 = kc * 128;
        if (kc < 3) {
            tdm_load_slice(wsrc + kb0 + 128, &btile[(kc + 1) & 1][wave * 8 * 128]);
            __builtin_amdgcn_s_wait_tensorcnt(1);   // chunk kc landed (in-order)
        } else {
            __builtin_amdgcn_s_wait_tensorcnt(0);
        }
        __syncthreads();                           // all waves' slices visible
        const bf16_t* bt = &btile[kc & 1][0];
        __builtin_prefetch(arow + kb0 + 128 + (size_t)(lane & 15) * EE, 0, 0);
#pragma unroll
        for (int ks = 0; ks < 4; ++ks) {           // K steps of 32
            int kb = kb0 + ks * 32;
            // batch all fragment loads first -> one clause, one wait, 4 WMMAs
            bf16x16 af = load_frag_a(arow + kb, EE, lane);
            bf16x16 bfv[4];
#pragma unroll
            for (int t = 0; t < 4; ++t)
                bfv[t] = load_frag_b(bt + (size_t)(t * 16) * 128 + ks * 32, 128, lane);
#pragma unroll
            for (int t = 0; t < 4; ++t)
                acc[t] = __builtin_amdgcn_wmma_f32_16x16x32_bf16(
                    false, af, false, bfv[t], (short)0, acc[t], false, false);
        }
        __syncthreads();                           // done reading before next DMA overwrite
    }

    int hi = (lane >> 4) * 8;
    int nl = lane & 15;
#pragma unroll
    for (int t = 0; t < 4; ++t) {
        int n = n0 + t * 16 + nl;
        float bv = bias[n];
        int h = n >> 6, f = n & 63;
#pragma unroll
        for (int r = 0; r < 8; ++r) {
            int m = m0 + hi + r;
            float val = acc[t][r] + bv;
            if (mode == 0) {
                ((bf16_t*)out)[(size_t)m * EE + n] = (bf16_t)val;
            } else if (mode == 1) {
                int b = m / TT, tt = m % TT;
                ((bf16_t*)out)[(((size_t)(b * HH + h)) * TT + tt) * 128 + f] = (bf16_t)val;
            } else if (mode == 2) {
                int b = m / TT, tt = m % TT;
                ((bf16_t*)out)[(((size_t)(b * HH + h)) * DHH + f) * TT + tt] = (bf16_t)val;
            } else {
                ((float*)out)[(size_t)m * EE + n] = val;
            }
        }
    }
}

// ---- kernel 5: build Qhat (q_u | trig-folded q_v) and Khat trig half ---
__global__ __launch_bounds__(256)
void hat_kernel(const bf16_t* __restrict__ Qtmp, const float* __restrict__ trig,
                const float* __restrict__ pbu, const float* __restrict__ pbv,
                bf16_t* __restrict__ Qhat, bf16_t* __restrict__ Khat) {
    int gid = blockIdx.x * blockDim.x + threadIdx.x;  // B*H*T*64 threads
    int f   = gid & 63;
    int bht = gid >> 6;
    int t   = bht % TT;
    int h   = (bht / TT) % HH;
    int b   = bht / (TT * HH);
    const bf16_t* qrow = Qtmp + ((size_t)(b * TT + t)) * EE + h * DHH;

    float qf = (float)qrow[f];
    Qhat[(size_t)bht * 128 + f] = (bf16_t)(qf + pbu[h * DHH + f]);

    int gi = f & 31;
    float qs = (float)qrow[gi]      + pbv[h * DHH + gi];        // pe sin half
    float qc = (float)qrow[32 + gi] + pbv[h * DHH + 32 + gi];   // pe cos half
    float ci = trig[t * 64 + gi];        // cos(t*w)
    float si = trig[t * 64 + 32 + gi];   // sin(t*w)
    float aq = (f < 32) ? (qs * si + qc * ci)      // pairs with cos(j*w)
                        : (-qs * ci + qc * si);    // pairs with sin(j*w)
    Qhat[(size_t)bht * 128 + 64 + f] = (bf16_t)aq;

    Khat[(size_t)bht * 128 + 64 + f] = (bf16_t)trig[t * 64 + f];
}

// ---- kernel 6: fused flash attention (scores never leave the WGP) ------
__global__ __launch_bounds__(256)
void attn_kernel(const bf16_t* __restrict__ Qhat, const bf16_t* __restrict__ Khat,
                 const bf16_t* __restrict__ Vt, const unsigned char* __restrict__ mask,
                 bf16_t* __restrict__ Obuf) {
    __shared__ bf16_t pshm[8][16 * 32];   // wave-private P staging (reshape C->A frag)
    int lane = threadIdx.x & 31;
    int wave = threadIdx.x >> 5;
    int bh   = blockIdx.x;                // b*H + h
    int b    = bh / HH;
    int h    = bh % HH;
    int i0   = blockIdx.y * 128 + wave * 16;

    const bf16_t* Qb = Qhat + ((size_t)bh * TT + i0) * 128;
    const bf16_t* Kb = Khat + (size_t)bh * TT * 128;
    const bf16_t* Vb = Vt   + (size_t)bh * DHH * TT;

    bf16x16 qf[4];
#pragma unroll
    for (int k = 0; k < 4; ++k) qf[k] = load_frag_a(Qb + k * 32, 128, lane);

    f32x8 z = {0.f,0.f,0.f,0.f,0.f,0.f,0.f,0.f};
    f32x8 o[4] = {z, z, z, z};
    float rmax[8], rsum[8];
#pragma unroll
    for (int r = 0; r < 8; ++r) { rmax[r] = -INFINITY; rsum[r] = 0.f; }
    const float scale = 0.125f;   // 1/sqrt(DH)

    for (int j0 = 0; j0 < TT; j0 += 32) {
        float sc[2][8];
#pragma unroll
        for (int half = 0; half < 2; ++half) {
            int jt = j0 + half * 16;
            // batch the 4 K-fragment loads, then chain 4 WMMAs
            bf16x16 kf[4];
#pragma unroll
            for (int k = 0; k < 4; ++k)
                kf[k] = load_frag_b(Kb + (size_t)jt * 128 + k * 32, 128, lane);
            f32x8 s = z;
#pragma unroll
            for (int k = 0; k < 4; ++k)
                s = __builtin_amdgcn_wmma_f32_16x16x32_bf16(
                    false, qf[k], false, kf[k], (short)0, s, false, false);
            int j = jt + (lane & 15);
            float madd = mask[b * TT + j] ? 0.f : -INFINITY;
#pragma unroll
            for (int r = 0; r < 8; ++r) sc[half][r] = (s[r] + madd) * scale;
        }

        // online softmax per row (row's 16 N-values live across one 16-lane half)
        float alpha[8];
#pragma unroll
        for (int r = 0; r < 8; ++r) {
            float mn = fmaxf(sc[0][r], sc[1][r]);
#pragma unroll
            for (int m = 1; m < 16; m <<= 1) mn = fmaxf(mn, __shfl_xor(mn, m, 32));
            mn = fmaxf(mn, rmax[r]);
            float al = __expf(rmax[r] - mn);
            rmax[r] = mn;
            alpha[r] = al;
            float p0 = __expf(sc[0][r] - mn);
            float p1 = __expf(sc[1][r] - mn);
            sc[0][r] = p0; sc[1][r] = p1;
            float ps = p0 + p1;
#pragma unroll
            for (int m = 1; m < 16; m <<= 1) ps += __shfl_xor(ps, m, 32);
            rsum[r] = rsum[r] * al + ps;
        }

        // stage P (16x32 bf16) in LDS, reload as an A fragment
        bf16_t* pw = &pshm[wave][0];
        int rowhi = (lane >> 4) * 8;
#pragma unroll
        for (int r = 0; r < 8; ++r) {
            int row = rowhi + r;
            pw[row * 32 + (lane & 15)]      = (bf16_t)sc[0][r];
            pw[row * 32 + 16 + (lane & 15)] = (bf16_t)sc[1][r];
        }
        bf16x16 pf = load_frag_a(pw, 32, lane);

        // batch V-fragment loads, rescale O, then 4 WMMAs
        bf16x16 vf[4];
#pragma unroll
        for (int t = 0; t < 4; ++t)
            vf[t] = load_frag_b(Vb + (size_t)(t * 16) * TT + j0, TT, lane);
#pragma unroll
        for (int t = 0; t < 4; ++t) {
#pragma unroll
            for (int r = 0; r < 8; ++r) o[t][r] *= alpha[r];
            o[t] = __builtin_amdgcn_wmma_f32_16x16x32_bf16(
                false, pf, false, vf[t], (short)0, o[t], false, false);
        }
    }

    int hi = (lane >> 4) * 8;
#pragma unroll
    for (int t = 0; t < 4; ++t) {
#pragma unroll
        for (int r = 0; r < 8; ++r) {
            int i = i0 + hi + r;
            int f = t * 16 + (lane & 15);
            Obuf[((size_t)(b * TT + i)) * EE + h * DHH + f] = (bf16_t)(o[t][r] / rsum[r]);
        }
    }
}

// ---- host orchestration ------------------------------------------------
extern "C" void kernel_launch(void* const* d_in, const int* in_sizes, int n_in,
                              void* d_out, int out_size, void* d_ws, size_t ws_size,
                              hipStream_t stream) {
    const float* x_in  = (const float*)d_in[0];
    const unsigned char* seqm = (const unsigned char*)d_in[1];
    const float* ln_g  = (const float*)d_in[2];
    const float* ln_b  = (const float*)d_in[3];
    const float* q_w   = (const float*)d_in[4];
    const float* k_w   = (const float*)d_in[5];
    const float* v_w   = (const float*)d_in[6];
    const float* ipb   = (const float*)d_in[7];   // [bias_k | bias_q | bias_v]
    const float* out_w = (const float*)d_in[8];
    const float* out_b = (const float*)d_in[9];
    const float* pbu   = (const float*)d_in[10];
    const float* pbv   = (const float*)d_in[11];

    char* ws = (char*)d_ws;
    size_t off = 0;
    bf16_t* xn   = (bf16_t*)(ws + off); off += (size_t)MM * EE * 2;            // 8 MB
    bf16_t* Qtmp = (bf16_t*)(ws + off); off += (size_t)MM * EE * 2;            // 8 MB
    bf16_t* Qhat = (bf16_t*)(ws + off); off += (size_t)BB * HH * TT * 128 * 2; // 16 MB
    bf16_t* Khat = (bf16_t*)(ws + off); off += (size_t)BB * HH * TT * 128 * 2; // 16 MB
    bf16_t* Vt   = (bf16_t*)(ws + off); off += (size_t)BB * HH * DHH * TT * 2; // 8 MB
    bf16_t* Obuf = (bf16_t*)(ws + off); off += (size_t)MM * EE * 2;            // 8 MB
    float*  trig = (float*) (ws + off); off += (size_t)TT * 64 * 4;            // 256 KB
    bf16_t* Wq   = (bf16_t*)(ws + off); off += (size_t)EE * EE * 2;
    bf16_t* Wk   = (bf16_t*)(ws + off); off += (size_t)EE * EE * 2;
    bf16_t* Wv   = (bf16_t*)(ws + off); off += (size_t)EE * EE * 2;
    bf16_t* Wo   = (bf16_t*)(ws + off); off += (size_t)EE * EE * 2;

    const int wn = EE * EE;
    f2bf_kernel<<<(wn + 255) / 256, 256, 0, stream>>>(q_w,   Wq, wn);
    f2bf_kernel<<<(wn + 255) / 256, 256, 0, stream>>>(k_w,   Wk, wn);
    f2bf_kernel<<<(wn + 255) / 256, 256, 0, stream>>>(v_w,   Wv, wn);
    f2bf_kernel<<<(wn + 255) / 256, 256, 0, stream>>>(out_w, Wo, wn);

    ln_kernel<<<MM / 8, 256, 0, stream>>>(x_in, ln_g, ln_b, xn);
    trig_kernel<<<(TT * 64) / 256, 256, 0, stream>>>(trig);

    dim3 ggrid(MM / 128, EE / 64);
    gemm_kernel<<<ggrid, 256, 0, stream>>>(xn, Wq, ipb + EE,     0, (void*)Qtmp);
    gemm_kernel<<<ggrid, 256, 0, stream>>>(xn, Wk, ipb,          1, (void*)Khat);
    gemm_kernel<<<ggrid, 256, 0, stream>>>(xn, Wv, ipb + 2 * EE, 2, (void*)Vt);

    hat_kernel<<<(BB * HH * TT * 64) / 256, 256, 0, stream>>>(Qtmp, trig, pbu, pbv, Qhat, Khat);

    dim3 agrid(BB * HH, TT / 128);
    attn_kernel<<<agrid, 256, 0, stream>>>(Qhat, Khat, Vt, seqm, Obuf);

    gemm_kernel<<<ggrid, 256, 0, stream>>>(Obuf, Wo, out_b, 3, d_out);
}
